// QuantSelfAttention_57878979281493
// MI455X (gfx1250) — matrix operand order repaired and checked by reading, not compiled, not run on hardware
//
#include <hip/hip_runtime.h>
#include <hip/hip_bf16.h>

typedef __attribute__((ext_vector_type(16))) _Float16 v16h;
typedef __attribute__((ext_vector_type(8)))  float    v8f;

#define B_ 4
#define H_ 16
#define L_ 1024
#define C_ 1024
#define D_ 64

// ---- WMMA fragment loaders (gfx1250 wave32 layouts, 16-bit operands) -------
// A (16xK=32): lane l, j=l>>4: v0..v3 = K[8j..8j+7], v4..v7 = K[16+8j..23+8j]
static __device__ __forceinline__ v16h frag_a(const _Float16* base, int ld,
                                              int row0, int k0, int lane) {
  int r = row0 + (lane & 15);
  int j = lane >> 4;
  const _Float16* p = base + (size_t)r * ld + k0 + j * 8;
  union { v16h h; uint4 q[2]; } u;
  u.q[0] = *reinterpret_cast<const uint4*>(p);
  u.q[1] = *reinterpret_cast<const uint4*>(p + 16);
  return u.h;
}

// B (K=32 x 16): lane l, col=l&15, j=l>>4: v0..v3 = K[8j..8j+7], v4..v7 = K[16+8j..23+8j]
static __device__ __forceinline__ v16h frag_b(const _Float16* base, int ld,
                                              int col0, int k0, int lane) {
  int n = col0 + (lane & 15);
  int j = lane >> 4;
  const _Float16* p = base + (size_t)n * ld + k0 + j * 8;
  union { v16h h; uint4 q[2]; } u;
  u.q[0] = *reinterpret_cast<const uint4*>(p);
  u.q[1] = *reinterpret_cast<const uint4*>(p + 16);
  return u.h;
}

static __device__ __forceinline__ v8f wmma_f16(v16h a, v16h b, v8f c) {
  return __builtin_amdgcn_wmma_f32_16x16x32_f16(false, a, false, b,
                                                (short)0, c, false, false);
}

// ---- f32 -> f16 convert ----------------------------------------------------
__global__ void k_cvt_f32_f16(const float* __restrict__ src,
                              _Float16* __restrict__ dst, int n) {
  int i = blockIdx.x * 256 + threadIdx.x;
  if (i < n) dst[i] = (_Float16)src[i];
}

// ---- QKV projection: [4096,1024] x [3072,1024]^T, 32x32 tile per wave -----
__global__ __launch_bounds__(256) void k_qkv_gemm(
    const _Float16* __restrict__ xh, const _Float16* __restrict__ wh,
    const float* __restrict__ q_bias, const float* __restrict__ v_bias,
    _Float16* __restrict__ qh, _Float16* __restrict__ kh,
    _Float16* __restrict__ vh) {
  int wave = threadIdx.x >> 5, lane = threadIdx.x & 31;
  int tile = blockIdx.x * 8 + wave;          // 128 x 96 tiles of 32x32
  int tr = tile / 96, tc = tile - tr * 96;
  int row0 = tr * 32, col0 = tc * 32;
  v8f acc00 = {0.f,0.f,0.f,0.f,0.f,0.f,0.f,0.f};
  v8f acc01 = acc00, acc10 = acc00, acc11 = acc00;
  for (int k0 = 0; k0 < C_; k0 += 32) {
    v16h a0 = frag_a(xh, C_, row0,      k0, lane);
    v16h a1 = frag_a(xh, C_, row0 + 16, k0, lane);
    v16h b0 = frag_b(wh, C_, col0,      k0, lane);
    v16h b1 = frag_b(wh, C_, col0 + 16, k0, lane);
    acc00 = wmma_f16(a0, b0, acc00);
    acc01 = wmma_f16(a0, b1, acc01);
    acc10 = wmma_f16(a1, b0, acc10);
    acc11 = wmma_f16(a1, b1, acc11);
  }
  int mofs = 8 * (lane >> 4);
#pragma unroll
  for (int ci = 0; ci < 2; ++ci) {
    int n = col0 + ci * 16 + (lane & 15);
    int s = n >> 10, r = n & 1023;
    int h = r >> 6, d = r & 63;
    float bias = (s == 0) ? q_bias[r] : (s == 2 ? v_bias[r] : 0.f);
    _Float16* dst = (s == 0) ? qh : (s == 1 ? kh : vh);
#pragma unroll
    for (int ri = 0; ri < 2; ++ri) {
      v8f acc = (ri == 0) ? (ci == 0 ? acc00 : acc01)
                          : (ci == 0 ? acc10 : acc11);
#pragma unroll
      for (int i = 0; i < 8; ++i) {
        int row = row0 + ri * 16 + i + mofs; // b*L + l
        int bb = row >> 10, l = row & 1023;
        dst[(((size_t)(bb * H_ + h)) * L_ + l) * D_ + d] =
            (_Float16)(acc[i] + bias);
      }
    }
  }
}

// ---- per-group (32 along D) symmetric 8-bit fake-quant of q and k ---------
__global__ __launch_bounds__(256) void k_quant_qk(_Float16* __restrict__ qh,
                                                  _Float16* __restrict__ kh) {
  int wave = threadIdx.x >> 5, lane = threadIdx.x & 31;
  int gid = blockIdx.x * 8 + wave;           // 0..131071
  int buf = gid >> 16;                       // 0 -> q, 1 -> k
  int row = gid & 65535;
  _Float16* p = (buf ? kh : qh) + (size_t)row * D_;
  float mult = buf ? 1.0f : 0.125f;          // fold D^-0.5 into quantized q
  float x0 = (float)p[lane];                 // group 0: d = 0..31
  float x1 = (float)p[lane + 32];            // group 1: d = 32..63
  float m0 = fabsf(x0), m1 = fabsf(x1);
#pragma unroll
  for (int off = 16; off; off >>= 1) {
    m0 = fmaxf(m0, __shfl_xor(m0, off, 32));
    m1 = fmaxf(m1, __shfl_xor(m1, off, 32));
  }
  float s0 = fmaxf(m0 * (1.f / 127.f), 1e-8f);
  float s1 = fmaxf(m1 * (1.f / 127.f), 1e-8f);
  float q0 = fminf(fmaxf(rintf(x0 / s0), -128.f), 127.f) * s0 * mult;
  float q1 = fminf(fmaxf(rintf(x1 / s1), -128.f), 127.f) * s1 * mult;
  p[lane] = (_Float16)q0;
  p[lane + 32] = (_Float16)q1;
}

// ---- per-(b,h,d) 8-bit fake-quant of v over L, write transposed [B,H,D,L] -
__global__ __launch_bounds__(256) void k_quant_v(const _Float16* __restrict__ vh,
                                                 _Float16* __restrict__ vth) {
  __shared__ float sv[L_];
  __shared__ float red[8];
  int tid = threadIdx.x, wave = tid >> 5, lane = tid & 31;
  int bhd = blockIdx.x;                      // 0..4095
  int d = bhd & 63, bh = bhd >> 6;
  const _Float16* src = vh + (size_t)bh * L_ * D_ + d;
  float m = 0.f;
  for (int l = tid; l < L_; l += 256) {
    float x = (float)src[(size_t)l * D_];
    sv[l] = x;
    m = fmaxf(m, fabsf(x));
  }
#pragma unroll
  for (int off = 16; off; off >>= 1) m = fmaxf(m, __shfl_xor(m, off, 32));
  if (lane == 0) red[wave] = m;
  __syncthreads();
  float mm = red[0];
#pragma unroll
  for (int w = 1; w < 8; ++w) mm = fmaxf(mm, red[w]);
  float s = fmaxf(mm * (1.f / 127.f), 1e-8f);
  _Float16* dst = vth + (size_t)bhd * L_;
  for (int l = tid; l < L_; l += 256) {
    float q = fminf(fmaxf(rintf(sv[l] / s), -128.f), 127.f) * s;
    dst[l] = (_Float16)q;
  }
}

// ---- fused attention per (b,h, 16-row block) ------------------------------
__global__ __launch_bounds__(256) void k_attn(
    const _Float16* __restrict__ qh, const _Float16* __restrict__ kh,
    const _Float16* __restrict__ vth, const float* __restrict__ attn_bias,
    _Float16* __restrict__ oh) {
  __shared__ float sc[16 * L_];              // 64 KB score tile
  int tid = threadIdx.x, wave = tid >> 5, lane = tid & 31;
  int bh = blockIdx.y;
  int r0 = blockIdx.x * 16;
  const _Float16* Q = qh + (size_t)bh * L_ * D_;
  const _Float16* K = kh + (size_t)bh * L_ * D_;
  const _Float16* VT = vth + (size_t)bh * D_ * L_;
  int mofs = 8 * (lane >> 4);

  // pass 1: S = (qq*scale) K^T + bias; Q fragments loop-invariant -> hoisted
  v16h aq0 = frag_a(Q, D_, r0, 0, lane);
  v16h aq1 = frag_a(Q, D_, r0, 32, lane);
  for (int ct = wave; ct < 64; ct += 8) {
    v8f acc = {0.f, 0.f, 0.f, 0.f, 0.f, 0.f, 0.f, 0.f};
    acc = wmma_f16(aq0, frag_b(K, D_, ct * 16, 0, lane), acc);
    acc = wmma_f16(aq1, frag_b(K, D_, ct * 16, 32, lane), acc);
    int n = ct * 16 + (lane & 15);
#pragma unroll
    for (int i = 0; i < 8; ++i) {
      int lr = i + mofs;
      sc[lr * L_ + n] = acc[i] + attn_bias[(size_t)(r0 + lr) * L_ + n];
    }
  }
  __syncthreads();

  // softmax + 8-bit static quant; prob row rr overlaid as f16 at byte rr*4096
  // (a wave only overwrites the float rows it itself reads -> race-free)
  _Float16* pr = reinterpret_cast<_Float16*>(sc);
  for (int rr = wave * 2; rr < wave * 2 + 2; ++rr) {
    float* row = sc + rr * L_;
    float m = -1e30f;
#pragma unroll
    for (int t = 0; t < 32; ++t) m = fmaxf(m, row[lane + t * 32]);
#pragma unroll
    for (int off = 16; off; off >>= 1) m = fmaxf(m, __shfl_xor(m, off, 32));
    float ev[32];
    float sum = 0.f;
#pragma unroll
    for (int t = 0; t < 32; ++t) {
      float e = __expf(row[lane + t * 32] - m);
      ev[t] = e;
      sum += e;
    }
#pragma unroll
    for (int off = 16; off; off >>= 1) sum += __shfl_xor(sum, off, 32);
    float inv = 1.f / sum;
#pragma unroll
    for (int t = 0; t < 32; ++t) {
      float pv = ev[t] * inv;
      pv = fminf(fmaxf(rintf(pv * 255.f), 0.f), 255.f) * (1.f / 255.f);
      pr[rr * 2048 + lane + t * 32] = (_Float16)pv;
    }
  }
  __syncthreads();

  // pass 2: O = P * V   (probs from LDS, VT contiguous along K)
  if (wave < 4) {
    v8f acc = {0.f, 0.f, 0.f, 0.f, 0.f, 0.f, 0.f, 0.f};
    int d0 = wave * 16;
    for (int k0 = 0; k0 < L_; k0 += 32) {
      v16h a = frag_a(pr, 2048, 0, k0, lane);
      v16h b = frag_b(VT, L_, d0, k0, lane);
      acc = wmma_f16(a, b, acc);
    }
    int b = bh >> 4, h = bh & 15;
    int n = d0 + (lane & 15);
#pragma unroll
    for (int i = 0; i < 8; ++i) {
      int l = r0 + i + mofs;
      oh[((size_t)(b * L_ + l)) * C_ + h * D_ + n] = (_Float16)acc[i];
    }
  }
}

// ---- output projection: [4096,1024] x [1024,1024]^T, 32x32 tile per wave --
__global__ __launch_bounds__(256) void k_proj(
    const _Float16* __restrict__ oh, const _Float16* __restrict__ wph,
    const float* __restrict__ b_proj, float* __restrict__ out) {
  int wave = threadIdx.x >> 5, lane = threadIdx.x & 31;
  int tile = blockIdx.x * 8 + wave;          // 128 x 32 tiles of 32x32
  int tr = tile >> 5, tc = tile & 31;
  int row0 = tr * 32, col0 = tc * 32;
  v8f acc00 = {0.f,0.f,0.f,0.f,0.f,0.f,0.f,0.f};
  v8f acc01 = acc00, acc10 = acc00, acc11 = acc00;
  for (int k0 = 0; k0 < C_; k0 += 32) {
    v16h a0 = frag_a(oh, C_, row0,      k0, lane);
    v16h a1 = frag_a(oh, C_, row0 + 16, k0, lane);
    v16h b0 = frag_b(wph, C_, col0,      k0, lane);
    v16h b1 = frag_b(wph, C_, col0 + 16, k0, lane);
    acc00 = wmma_f16(a0, b0, acc00);
    acc01 = wmma_f16(a0, b1, acc01);
    acc10 = wmma_f16(a1, b0, acc10);
    acc11 = wmma_f16(a1, b1, acc11);
  }
  int mofs = 8 * (lane >> 4);
#pragma unroll
  for (int ci = 0; ci < 2; ++ci) {
    int n = col0 + ci * 16 + (lane & 15);
    float bias = b_proj[n];
#pragma unroll
    for (int ri = 0; ri < 2; ++ri) {
      v8f acc = (ri == 0) ? (ci == 0 ? acc00 : acc01)
                          : (ci == 0 ? acc10 : acc11);
#pragma unroll
      for (int i = 0; i < 8; ++i) {
        out[(size_t)(row0 + ri * 16 + i + mofs) * C_ + n] = acc[i] + bias;
      }
    }
  }
}

extern "C" void kernel_launch(void* const* d_in, const int* in_sizes, int n_in,
                              void* d_out, int out_size, void* d_ws,
                              size_t ws_size, hipStream_t stream) {
  (void)in_sizes; (void)n_in; (void)out_size; (void)ws_size;
  const float* x         = (const float*)d_in[0];
  const float* attn_bias = (const float*)d_in[1];
  const float* w_qkv     = (const float*)d_in[2];
  const float* q_bias    = (const float*)d_in[3];
  const float* v_bias    = (const float*)d_in[4];
  const float* w_proj    = (const float*)d_in[5];
  const float* b_proj    = (const float*)d_in[6];
  float* out = (float*)d_out;

  // workspace carve-up (f16 buffers), total ~56 MB
  char* ws = (char*)d_ws;
  _Float16* xh  = (_Float16*)ws; ws += (size_t)B_ * L_ * C_ * 2;       // 8 MB
  _Float16* wqh = (_Float16*)ws; ws += (size_t)3 * C_ * C_ * 2;        // 6 MB
  _Float16* wph = (_Float16*)ws; ws += (size_t)C_ * C_ * 2;            // 2 MB
  _Float16* qh  = (_Float16*)ws; ws += (size_t)B_ * H_ * L_ * D_ * 2;  // 8 MB
  _Float16* kh  = (_Float16*)ws; ws += (size_t)B_ * H_ * L_ * D_ * 2;  // 8 MB
  _Float16* vh  = (_Float16*)ws; ws += (size_t)B_ * H_ * L_ * D_ * 2;  // 8 MB
  _Float16* vth = (_Float16*)ws; ws += (size_t)B_ * H_ * D_ * L_ * 2;  // 8 MB
  _Float16* oh  = (_Float16*)ws; ws += (size_t)B_ * L_ * C_ * 2;       // 8 MB

  k_cvt_f32_f16<<<(B_ * L_ * C_ + 255) / 256, 256, 0, stream>>>(x, xh, B_ * L_ * C_);
  k_cvt_f32_f16<<<(3 * C_ * C_ + 255) / 256, 256, 0, stream>>>(w_qkv, wqh, 3 * C_ * C_);
  k_cvt_f32_f16<<<(C_ * C_ + 255) / 256, 256, 0, stream>>>(w_proj, wph, C_ * C_);

  // 128*96 = 12288 tiles, 8 waves/block
  k_qkv_gemm<<<1536, 256, 0, stream>>>(xh, wqh, q_bias, v_bias, qh, kh, vh);
  k_quant_qk<<<16384, 256, 0, stream>>>(qh, kh);
  k_quant_v<<<4096, 256, 0, stream>>>(vh, vth);

  dim3 ga(L_ / 16, B_ * H_);
  k_attn<<<ga, 256, 0, stream>>>(qh, kh, vth, attn_bias, oh);

  // 128*32 = 4096 tiles, 8 waves/block
  k_proj<<<512, 256, 0, stream>>>(oh, wph, b_proj, out);
}
